// GroupTransformer_78254304133454
// MI455X (gfx1250) — matrix operand order repaired and checked by reading, not compiled, not run on hardware
//
#include <hip/hip_runtime.h>

// ---------------------------------------------------------------------------
// GroupTransformer for MI455X (gfx1250, wave32, WMMA).
// bf16 storage for matmul operands, f32 accumulation via
// v_wmma_f32_16x16x32_bf16. Each wave computes a 16x64 output strip (four
// independent accumulators) so the 4-slot WMMA->WMMA hazard window is filled
// with real WMMAs instead of v_nops, and the A fragment is reused 4x.
// LDS/global f32 atomics for the one-hot scatter, LDS tree reductions for
// softmax / L2 norms. Intermediates that only feed matmuls are bf16.
// ---------------------------------------------------------------------------

typedef __attribute__((ext_vector_type(16))) __bf16 v16bf;
typedef __attribute__((ext_vector_type(8)))  float  v8f;

union V16 {
  v16bf v;
  unsigned short us[16];
  uint4 q[2];
};

__device__ __forceinline__ unsigned short f2bf(float f) {
  unsigned u = __float_as_uint(f);
  u += 0x7fffu + ((u >> 16) & 1u);   // round-to-nearest-even
  return (unsigned short)(u >> 16);
}
__device__ __forceinline__ float bf2f(unsigned short h) {
  return __uint_as_float(((unsigned)h) << 16);
}

__device__ __forceinline__ float block_sum256(float v, float* red) {
  int t = threadIdx.x;
  red[t] = v; __syncthreads();
#pragma unroll
  for (int s = 128; s > 0; s >>= 1) {
    if (t < s) red[t] += red[t + s];
    __syncthreads();
  }
  float r = red[0]; __syncthreads();
  return r;
}
__device__ __forceinline__ float block_max256(float v, float* red) {
  int t = threadIdx.x;
  red[t] = v; __syncthreads();
#pragma unroll
  for (int s = 128; s > 0; s >>= 1) {
    if (t < s) red[t] = fmaxf(red[t], red[t + s]);
    __syncthreads();
  }
  float r = red[0]; __syncthreads();
  return r;
}

// One 16x64 strip: bf16 A row-major [*, lda]; Bt row-major [*, ldb]
// (Bt[n,k] = B[k,n]); Arow/Brow* pre-offset to this lane's rows.
// Four independent accumulators -> WMMA self-dependence is 4 back.
__device__ __forceinline__ void wmma_strip4(
    const unsigned short* __restrict__ Arow,
    const unsigned short* __restrict__ Brow0,
    const unsigned short* __restrict__ Brow1,
    const unsigned short* __restrict__ Brow2,
    const unsigned short* __restrict__ Brow3,
    int K, int koffA, int koffB,
    v8f& acc0, v8f& acc1, v8f& acc2, v8f& acc3) {
  for (int kk = 0; kk < K; kk += 32) {
    V16 a, b0, b1, b2, b3;
    a.q[0]  = *(const uint4*)(Arow + kk + koffA);
    a.q[1]  = *(const uint4*)(Arow + kk + koffA + 8);
    b0.q[0] = *(const uint4*)(Brow0 + kk + koffB);
    b0.q[1] = *(const uint4*)(Brow0 + kk + koffB + 8);
    b1.q[0] = *(const uint4*)(Brow1 + kk + koffB);
    b1.q[1] = *(const uint4*)(Brow1 + kk + koffB + 8);
    b2.q[0] = *(const uint4*)(Brow2 + kk + koffB);
    b2.q[1] = *(const uint4*)(Brow2 + kk + koffB + 8);
    b3.q[0] = *(const uint4*)(Brow3 + kk + koffB);
    b3.q[1] = *(const uint4*)(Brow3 + kk + koffB + 8);
    acc0 = __builtin_amdgcn_wmma_f32_16x16x32_bf16(
        false, a.v, false, b0.v, (short)0, acc0, false, false);
    acc1 = __builtin_amdgcn_wmma_f32_16x16x32_bf16(
        false, a.v, false, b1.v, (short)0, acc1, false, false);
    acc2 = __builtin_amdgcn_wmma_f32_16x16x32_bf16(
        false, a.v, false, b2.v, (short)0, acc2, false, false);
    acc3 = __builtin_amdgcn_wmma_f32_16x16x32_bf16(
        false, a.v, false, b3.v, (short)0, acc3, false, false);
  }
}

// Epilogue modes
constexpr int EP_XP   = 0;  // + bias + pos_emb[row%4096], store bf16
constexpr int EP_BF16 = 1;  // + bias, store bf16
constexpr int EP_F32  = 2;  // + bias, store f32
constexpr int EP_GELU = 3;  // gelu(exact)(acc+bias), store bf16

// Generic GEMM: C[M,Nc] = A[M,K] @ Bt[Nc,K]^T (+epilogue). One wave per
// 16x64 strip; Nc must be a multiple of 64, M of 16, K of 32.
template <int MODE, bool AF32>
__global__ __launch_bounds__(256) void gemm_kernel(
    const void* __restrict__ Ap, int lda,
    const unsigned short* __restrict__ Bt, int ldb,
    void* __restrict__ Cp, int ldc,
    const float* __restrict__ bias, const float* __restrict__ pos,
    int M, int Nc, int K) {
  int lane = threadIdx.x & 31;
  int w    = (blockIdx.x * 256 + threadIdx.x) >> 5;
  int tnq  = Nc >> 6;                 // 64-wide quads along N
  int tile_m = w / tnq;
  int n0     = (w - tile_m * tnq) << 6;
  if (tile_m * 16 >= M) return;

  int mr    = lane & 15;
  int koffA = (lane < 16) ? 0 : 16;
  int koffB = (lane < 16) ? 0 : 16;
  const unsigned short* Brow0 = Bt + (size_t)(n0 + mr) * ldb;
  const unsigned short* Brow1 = Brow0 + (size_t)16 * ldb;
  const unsigned short* Brow2 = Brow0 + (size_t)32 * ldb;
  const unsigned short* Brow3 = Brow0 + (size_t)48 * ldb;

  v8f acc[4];
#pragma unroll
  for (int j = 0; j < 4; j++) acc[j] = (v8f){0.f,0.f,0.f,0.f,0.f,0.f,0.f,0.f};

  if constexpr (AF32) {
    const float* Arow = (const float*)Ap + (size_t)(tile_m * 16 + mr) * lda;
    for (int kk = 0; kk < K; kk += 32) {
      V16 a, b0, b1, b2, b3;
      float4 f0 = *(const float4*)(Arow + kk + koffA);
      float4 f1 = *(const float4*)(Arow + kk + koffA + 4);
      float4 f2 = *(const float4*)(Arow + kk + koffA + 8);
      float4 f3 = *(const float4*)(Arow + kk + koffA + 12);
      a.us[0]  = f2bf(f0.x); a.us[1]  = f2bf(f0.y);
      a.us[2]  = f2bf(f0.z); a.us[3]  = f2bf(f0.w);
      a.us[4]  = f2bf(f1.x); a.us[5]  = f2bf(f1.y);
      a.us[6]  = f2bf(f1.z); a.us[7]  = f2bf(f1.w);
      a.us[8]  = f2bf(f2.x); a.us[9]  = f2bf(f2.y);
      a.us[10] = f2bf(f2.z); a.us[11] = f2bf(f2.w);
      a.us[12] = f2bf(f3.x); a.us[13] = f2bf(f3.y);
      a.us[14] = f2bf(f3.z); a.us[15] = f2bf(f3.w);
      b0.q[0] = *(const uint4*)(Brow0 + kk + koffB);
      b0.q[1] = *(const uint4*)(Brow0 + kk + koffB + 8);
      b1.q[0] = *(const uint4*)(Brow1 + kk + koffB);
      b1.q[1] = *(const uint4*)(Brow1 + kk + koffB + 8);
      b2.q[0] = *(const uint4*)(Brow2 + kk + koffB);
      b2.q[1] = *(const uint4*)(Brow2 + kk + koffB + 8);
      b3.q[0] = *(const uint4*)(Brow3 + kk + koffB);
      b3.q[1] = *(const uint4*)(Brow3 + kk + koffB + 8);
      acc[0] = __builtin_amdgcn_wmma_f32_16x16x32_bf16(
          false, a.v, false, b0.v, (short)0, acc[0], false, false);
      acc[1] = __builtin_amdgcn_wmma_f32_16x16x32_bf16(
          false, a.v, false, b1.v, (short)0, acc[1], false, false);
      acc[2] = __builtin_amdgcn_wmma_f32_16x16x32_bf16(
          false, a.v, false, b2.v, (short)0, acc[2], false, false);
      acc[3] = __builtin_amdgcn_wmma_f32_16x16x32_bf16(
          false, a.v, false, b3.v, (short)0, acc[3], false, false);
    }
  } else {
    const unsigned short* Arow =
        (const unsigned short*)Ap + (size_t)(tile_m * 16 + mr) * lda;
    wmma_strip4(Arow, Brow0, Brow1, Brow2, Brow3, K, koffA, koffB,
                acc[0], acc[1], acc[2], acc[3]);
  }

  int rbase = (lane < 16) ? 0 : 8;
#pragma unroll
  for (int j = 0; j < 4; j++) {
    int col = n0 + j * 16 + (lane & 15);
    float bcol = bias[col];
#pragma unroll
    for (int r = 0; r < 8; r++) {
      int row  = tile_m * 16 + rbase + r;
      float vv = acc[j][r] + bcol;
      if constexpr (MODE == EP_XP) {
        vv += pos[(size_t)(row & 4095) * 256 + col];
        ((unsigned short*)Cp)[(size_t)row * ldc + col] = f2bf(vv);
      } else if constexpr (MODE == EP_BF16) {
        ((unsigned short*)Cp)[(size_t)row * ldc + col] = f2bf(vv);
      } else if constexpr (MODE == EP_F32) {
        ((float*)Cp)[(size_t)row * ldc + col] = vv;
      } else {  // EP_GELU, exact: 0.5*x*(1+erf(x/sqrt(2)))
        float g = 0.5f * vv * (1.0f + erff(vv * 0.7071067811865475f));
        ((unsigned short*)Cp)[(size_t)row * ldc + col] = f2bf(g);
      }
    }
  }
}

// scores[n,h,g,l] = (q[h,g,:] . k[n,h,l,:]) / 8 ; written into the soft output
// region (later softmaxed in place). qb: bf16 [G=64,256]; kb: bf16 [N*L,256].
// One wave per 16(g) x 64(l) strip; 256 waves per (n,h) batch.
__global__ __launch_bounds__(256) void scores_kernel(
    const unsigned short* __restrict__ qb,
    const unsigned short* __restrict__ kb,
    float* __restrict__ out) {
  int lane = threadIdx.x & 31;
  int w    = (blockIdx.x * 256 + threadIdx.x) >> 5;  // 0..255 per batch
  int b    = blockIdx.y;                             // n*H + h, 0..63
  int n = b >> 2, h = b & 3;
  int tile_m = w >> 6;          // G tiles: 0..3
  int l0     = (w & 63) << 6;   // L strip start
  int mr    = lane & 15;
  int koffA = (lane < 16) ? 0 : 16;
  int koffB = (lane < 16) ? 0 : 16;
  const unsigned short* Arow = qb + h * 64 + (size_t)(tile_m * 16 + mr) * 256;
  const unsigned short* Brow0 =
      kb + ((size_t)n * 4096 + l0 + mr) * 256 + h * 64;
  const unsigned short* Brow1 = Brow0 + (size_t)16 * 256;
  const unsigned short* Brow2 = Brow0 + (size_t)32 * 256;
  const unsigned short* Brow3 = Brow0 + (size_t)48 * 256;

  v8f acc[4];
#pragma unroll
  for (int j = 0; j < 4; j++) acc[j] = (v8f){0.f,0.f,0.f,0.f,0.f,0.f,0.f,0.f};
  wmma_strip4(Arow, Brow0, Brow1, Brow2, Brow3, 64, koffA, koffB,
              acc[0], acc[1], acc[2], acc[3]);

  int rbase = (lane < 16) ? 0 : 8;
  float* Cb = out + (size_t)b * 64 * 4096;
#pragma unroll
  for (int j = 0; j < 4; j++) {
    int col = l0 + j * 16 + (lane & 15);
#pragma unroll
    for (int r = 0; r < 8; r++) {
      int g = tile_m * 16 + rbase + r;
      Cb[(size_t)g * 4096 + col] = acc[j][r] * 0.125f;
    }
  }
}

// In-place softmax over rows of 4096 (rows = N*H*G = 4096).
__global__ __launch_bounds__(256) void softmax_kernel(float* __restrict__ soft) {
  __shared__ float red[256];
  float* p = soft + (size_t)blockIdx.x * 4096;
  int t = threadIdx.x;
  float vals[16];
  float mx = -3.402823466e38f;
#pragma unroll
  for (int i = 0; i < 16; i++) {
    vals[i] = p[t + i * 256];
    mx = fmaxf(mx, vals[i]);
  }
  mx = block_max256(mx, red);
  float s = 0.f;
#pragma unroll
  for (int i = 0; i < 16; i++) {
    vals[i] = expf(vals[i] - mx);
    s += vals[i];
  }
  s = block_sum256(s, red);
  float inv = 1.0f / s;
#pragma unroll
  for (int i = 0; i < 16; i++) p[t + i * 256] = vals[i] * inv;
}

// argmax over G (first max wins, like jnp.argmax) -> one-hot `hard` + assign.
__global__ __launch_bounds__(256) void hard_kernel(
    const float* __restrict__ soft, float* __restrict__ hard,
    int* __restrict__ assign) {
  int idx = blockIdx.x * 256 + threadIdx.x;  // (n*H+h)*L + l, 0..262143
  int nh = idx >> 12;
  int l  = idx & 4095;
  const float* sp = soft + (size_t)nh * (64 * 4096) + l;
  float best = sp[0];
  int bi = 0;
  for (int g = 1; g < 64; g++) {
    float s = sp[(size_t)g * 4096];
    if (s > best) { best = s; bi = g; }
  }
  float* hp = hard + (size_t)nh * (64 * 4096) + l;
  for (int g = 0; g < 64; g++) hp[(size_t)g * 4096] = (g == bi) ? 1.0f : 0.0f;
  assign[idx] = bi;
}

__global__ __launch_bounds__(256) void zero_kernel(float* __restrict__ p, int n) {
  int i = blockIdx.x * 256 + threadIdx.x;
  if (i < n) p[i] = 0.f;
}

// agg[n,h,g,dh] += v[n,l,h*64+dh] where assign[n,h,l]==g.
// LDS 64x64 f32 tile per (n,h), global f32 atomics to merge l-chunks.
__global__ __launch_bounds__(256) void agg_kernel(
    const unsigned short* __restrict__ vb, const int* __restrict__ assign,
    float* __restrict__ agg) {
  __shared__ float sagg[64 * 64];
  int t = threadIdx.x;
  for (int i = t; i < 4096; i += 256) sagg[i] = 0.f;
  __syncthreads();
  int nh = blockIdx.y;  // 0..63
  int n = nh >> 2, h = nh & 3;
  int lbase = blockIdx.x * 1024;  // gridDim.x = 4
  int dlane = t & 63, lg = t >> 6;
  for (int lo = 0; lo < 256; lo++) {
    int l = lbase + lo * 4 + lg;
    int a = assign[(size_t)nh * 4096 + l];
    float val = bf2f(vb[((size_t)n * 4096 + l) * 256 + h * 64 + dlane]);
    atomicAdd(&sagg[a * 64 + dlane], val);
  }
  __syncthreads();
  float* ag = agg + (size_t)nh * 4096;
  for (int i = t; i < 4096; i += 256) atomicAdd(&ag[i], sagg[i]);
}

// l2-normalize merged agg rows [N*G, 256] -> bf16 for the Wo GEMM.
__global__ __launch_bounds__(256) void aggnorm_kernel(
    const float* __restrict__ agg, unsigned short* __restrict__ aggn) {
  __shared__ float red[256];
  int row = blockIdx.x;  // n*64+g
  int n = row >> 6, g = row & 63;
  int d = threadIdx.x, h = d >> 6, dh = d & 63;
  float v = agg[(((size_t)n * 4 + h) * 64 + g) * 64 + dh];
  float ss = block_sum256(v * v, red);
  float sc = rsqrtf(fmaxf(ss, 1e-12f));
  aggn[(size_t)row * 256 + d] = f2bf(v * sc);
}

// t = group_emb(broadcast) + aggo ; accumulate global sum-of-squares.
__global__ __launch_bounds__(256) void gfsum_kernel(
    const float* __restrict__ aggo, const float* __restrict__ gemb,
    float* __restrict__ t, float* __restrict__ ssq) {
  __shared__ float red[256];
  int idx = blockIdx.x * 256 + threadIdx.x;  // 0..262143
  float v = gemb[idx & 16383] + aggo[idx];
  t[idx] = v;
  float s = block_sum256(v * v, red);
  if (threadIdx.x == 0) atomicAdd(ssq, s);
}

__global__ __launch_bounds__(256) void gffin_kernel(
    const float* __restrict__ t, const float* __restrict__ ssq,
    float* __restrict__ gf, unsigned short* __restrict__ gfb) {
  int idx = blockIdx.x * 256 + threadIdx.x;
  float sc = rsqrtf(fmaxf(ssq[0], 1e-12f));
  float v = t[idx] * sc;
  gf[idx] = v;
  gfb[idx] = f2bf(v);
}

__global__ __launch_bounds__(256) void final_kernel(
    const float* __restrict__ gf, const float* __restrict__ mlp,
    float* __restrict__ rslt) {
  __shared__ float red[256];
  int row = blockIdx.x, d = threadIdx.x;
  float v = gf[(size_t)row * 256 + d] + mlp[(size_t)row * 256 + d];
  float ss = block_sum256(v * v, red);
  float sc = rsqrtf(fmaxf(ss, 1e-12f));
  rslt[(size_t)row * 256 + d] = v * sc;
}

// Convert 7 weight matrices fp32 [256,256] -> bf16 transposed WT[c*256+r],
// and group_emb [64,256] -> bf16 (row major).
__global__ __launch_bounds__(256) void prep_kernel(
    const float* __restrict__ Wi, const float* __restrict__ Wq,
    const float* __restrict__ Wk, const float* __restrict__ Wv,
    const float* __restrict__ Wo, const float* __restrict__ W1,
    const float* __restrict__ W2, const float* __restrict__ gemb,
    unsigned short* __restrict__ WT, unsigned short* __restrict__ geb) {
  int which = blockIdx.y;
  int idx = blockIdx.x * 256 + threadIdx.x;
  if (which == 7) {
    if (idx < 64 * 256) geb[idx] = f2bf(gemb[idx]);
    return;
  }
  const float* src = (which == 0) ? Wi
                   : (which == 1) ? Wq
                   : (which == 2) ? Wk
                   : (which == 3) ? Wv
                   : (which == 4) ? Wo
                   : (which == 5) ? W1 : W2;
  int r = idx >> 8, c = idx & 255;
  WT[(size_t)which * 65536 + (size_t)c * 256 + r] = f2bf(src[idx]);
}

// ---------------------------------------------------------------------------

extern "C" void kernel_launch(void* const* d_in, const int* in_sizes, int n_in,
                              void* d_out, int out_size, void* d_ws,
                              size_t ws_size, hipStream_t stream) {
  (void)in_sizes; (void)n_in; (void)out_size; (void)ws_size;
  const float* x    = (const float*)d_in[0];
  const float* Wi   = (const float*)d_in[1];
  const float* bi   = (const float*)d_in[2];
  const float* pos  = (const float*)d_in[3];
  const float* gemb = (const float*)d_in[4];
  const float* Wq   = (const float*)d_in[5];
  const float* bq   = (const float*)d_in[6];
  const float* Wk   = (const float*)d_in[7];
  const float* bk   = (const float*)d_in[8];
  const float* Wv   = (const float*)d_in[9];
  const float* bv   = (const float*)d_in[10];
  const float* Wo   = (const float*)d_in[11];
  const float* bo   = (const float*)d_in[12];
  const float* W1   = (const float*)d_in[13];
  const float* b1   = (const float*)d_in[14];
  const float* W2   = (const float*)d_in[15];
  const float* b2   = (const float*)d_in[16];

  // Output layout: rslt [16,64,256] | soft [16,4,64,4096] | hard [same]
  float* rslt = (float*)d_out;
  float* soft = rslt + 262144;
  float* hard = soft + 16777216;

  // Workspace carve-up (all sizes multiples of 256B).
  char* ws = (char*)d_ws;
  size_t off = 0;
  auto carve = [&](size_t bytes) {
    char* p = ws + off;
    off += (bytes + 255) & ~(size_t)255;
    return (void*)p;
  };
  unsigned short* WT   = (unsigned short*)carve(7 * 65536 * 2);   // 7 bf16 Wt
  unsigned short* geb  = (unsigned short*)carve(64 * 256 * 2);    // group_emb bf16
  unsigned short* qb   = (unsigned short*)carve(64 * 256 * 2);    // q bf16 [G,D]
  unsigned short* xp   = (unsigned short*)carve((size_t)65536 * 256 * 2);
  unsigned short* kb   = (unsigned short*)carve((size_t)65536 * 256 * 2);
  unsigned short* vb   = (unsigned short*)carve((size_t)65536 * 256 * 2);
  float*          agg  = (float*)carve(262144 * 4);               // [N,H,G,64]
  float*          ssq  = (float*)carve(256);                      // global sumsq
  int*            asgn = (int*)carve(262144 * 4);                 // [N,H,L]
  unsigned short* aggn = (unsigned short*)carve(1024 * 256 * 2);
  float*          aggo = (float*)carve(262144 * 4);
  float*          gft  = (float*)carve(262144 * 4);
  float*          gf   = (float*)carve(262144 * 4);
  unsigned short* gfb  = (unsigned short*)carve(1024 * 256 * 2);
  unsigned short* h1b  = (unsigned short*)carve(1024 * 256 * 2);
  float*          mlp  = (float*)carve(262144 * 4);

  // 0) weight transpose + bf16 conversions
  prep_kernel<<<dim3(256, 8), 256, 0, stream>>>(Wi, Wq, Wk, Wv, Wo, W1, W2,
                                                gemb, WT, geb);

  // 1) xp = x@Wi + bi + pos_emb  (A fp32, converted inline)  [65536x256]
  //    waves = (65536/16)*(256/64) = 16384 -> 2048 blocks
  gemm_kernel<EP_XP, true><<<2048, 256, 0, stream>>>(
      x, 256, WT + 0 * 65536, 256, xp, 256, bi, pos, 65536, 256, 256);

  // 2) k = xp@Wk + bk (bf16) ; v = xp@Wv + bv (bf16)
  gemm_kernel<EP_BF16, false><<<2048, 256, 0, stream>>>(
      xp, 256, WT + 2 * 65536, 256, kb, 256, bk, nullptr, 65536, 256, 256);
  gemm_kernel<EP_BF16, false><<<2048, 256, 0, stream>>>(
      xp, 256, WT + 3 * 65536, 256, vb, 256, bv, nullptr, 65536, 256, 256);

  // 3) q = group_emb@Wq + bq (bf16, [64,256]) : 16 waves -> 2 blocks
  gemm_kernel<EP_BF16, false><<<2, 256, 0, stream>>>(
      geb, 256, WT + 1 * 65536, 256, qb, 256, bq, nullptr, 64, 256, 256);

  // 4) raw scores into the soft output region, then softmax in place
  //    256 waves per (n,h) batch -> 32 blocks x 64 batches
  scores_kernel<<<dim3(32, 64), 256, 0, stream>>>(qb, kb, soft);
  softmax_kernel<<<4096, 256, 0, stream>>>(soft);

  // 5) hard one-hot (forward value of the STE) + group assignment
  hard_kernel<<<1024, 256, 0, stream>>>(soft, hard, asgn);

  // 6) zero agg + ssq (adjacent), then scatter-sum v per group
  zero_kernel<<<1025, 256, 0, stream>>>(agg, 262144 + 64);
  agg_kernel<<<dim3(4, 64), 256, 0, stream>>>(vb, asgn, agg);

  // 7) row-l2n(agg) -> bf16 ; aggo = aggn@Wo + bo (f32): 256 waves -> 32 blocks
  aggnorm_kernel<<<1024, 256, 0, stream>>>(agg, aggn);
  gemm_kernel<EP_F32, false><<<32, 256, 0, stream>>>(
      aggn, 256, WT + 4 * 65536, 256, aggo, 256, bo, nullptr, 1024, 256, 256);

  // 8) gf = global-l2n(group_emb + aggo)
  gfsum_kernel<<<1024, 256, 0, stream>>>(aggo, gemb, gft, ssq);
  gffin_kernel<<<1024, 256, 0, stream>>>(gft, ssq, gf, gfb);

  // 9) mlp = gelu(gf@W1+b1)@W2 + b2
  gemm_kernel<EP_GELU, false><<<32, 256, 0, stream>>>(
      gfb, 256, WT + 5 * 65536, 256, h1b, 256, b1, nullptr, 1024, 256, 256);
  gemm_kernel<EP_F32, false><<<32, 256, 0, stream>>>(
      h1b, 256, WT + 6 * 65536, 256, mlp, 256, b2, nullptr, 1024, 256, 256);

  // 10) rslt = row-l2n(gf + mlp)
  final_kernel<<<1024, 256, 0, stream>>>(gf, mlp, rslt);
}
